// TransformBlock_45947560133458
// MI455X (gfx1250) — compile-verified, matrix operand order checked by reading, not run on hardware
//
#include <hip/hip_runtime.h>

// ---------------------------------------------------------------- constants
#define E_   1024
#define H_   16
#define HS_  64
#define B_   2
#define T_   2048
#define BT_  (B_ * T_)
#define FF_  (4 * E_)

// CDNA5 data-movement experiments (fallbacks preserved if assembler rejects)
#define USE_TDM   1   // tensor_load_to_lds for GEMM B tiles
#define USE_ASYNC 1   // global_load_async_to_lds_b128 for GEMM A tiles

typedef __attribute__((ext_vector_type(16))) __bf16     v16bf;
typedef __attribute__((ext_vector_type(8)))  float      v8f;
typedef __attribute__((ext_vector_type(4)))  unsigned   v4u;
typedef __attribute__((ext_vector_type(8)))  unsigned   v8u;

union Frag { v16bf v; uint4 q[2]; unsigned short us[16]; };

__device__ inline unsigned short f2bf(float f) {
  unsigned u = __float_as_uint(f);
  u += 0x7fffu + ((u >> 16) & 1u);           // round-to-nearest-even
  return (unsigned short)(u >> 16);
}

__device__ inline v8f vzero8() {
  v8f z = {0.f, 0.f, 0.f, 0.f, 0.f, 0.f, 0.f, 0.f};
  return z;
}

__device__ inline v8f wmma_bf16(const Frag& a, const Frag& b, v8f c) {
  return __builtin_amdgcn_wmma_f32_16x16x32_bf16(
      /*neg_a=*/false, a.v, /*neg_b=*/false, b.v,
      /*c_mod=*/(short)0, c, /*reuse_a=*/false, /*reuse_b=*/false);
}

// LDS offset of a shared-memory pointer: generic LDS addresses keep the
// LDS byte offset in the low 32 bits (ISA 10.2 aperture rules).
__device__ inline unsigned lds_off(const void* p) {
  return (unsigned)(unsigned long long)p;
}

__device__ inline void wait_async0() {
  asm volatile("s_wait_asynccnt 0x0" ::: "memory");
}

__device__ inline void wait_tensor0() {
#if __has_builtin(__builtin_amdgcn_s_wait_tensorcnt)
  __builtin_amdgcn_s_wait_tensorcnt(0);
#else
  asm volatile("s_wait_tensorcnt 0x0" ::: "memory");
#endif
}

// async copy 16 bytes global -> LDS (per-lane scatter, ASYNCcnt tracked)
__device__ inline void async_g2l_b128(unsigned loff, const void* g) {
  asm volatile("global_load_async_to_lds_b128 %0, %1, off"
               :: "v"(loff), "v"(g)
               : "memory");
}

// ------------------------------------------------- weight / dtype conversion
__global__ __launch_bounds__(256) void f32_to_bf16_kernel(
    const float* __restrict__ in, unsigned short* __restrict__ out, int n) {
  int i = blockIdx.x * 256 + threadIdx.x;
  if (i < n) out[i] = f2bf(in[i]);
}

// Wk/Wv: [H][E][HS] f32  ->  [E][H*HS] bf16  (so projection is one GEMM)
__global__ __launch_bounds__(256) void headperm_bf16_kernel(
    const float* __restrict__ in, unsigned short* __restrict__ out) {
  int idx = blockIdx.x * 256 + threadIdx.x;     // idx = e*1024 + h*64 + d
  int e = idx >> 10;
  int c = idx & 1023;
  int h = c >> 6;
  int d = c & 63;
  out[idx] = f2bf(in[(size_t)h * (E_ * HS_) + (size_t)e * HS_ + d]);
}

// ------------------------------------------------------------- layernorm
__global__ __launch_bounds__(256) void layernorm_bf16_kernel(
    const float* __restrict__ x, const float* __restrict__ g,
    const float* __restrict__ b, unsigned short* __restrict__ out) {
  __shared__ float red[16];
  const int row = blockIdx.x;
  const int tid = threadIdx.x;
  const int lane = tid & 31, w = tid >> 5;

  float4 xv = ((const float4*)x)[(size_t)row * (E_ / 4) + tid];
  float s  = xv.x + xv.y + xv.z + xv.w;
  float s2 = xv.x * xv.x + xv.y * xv.y + xv.z * xv.z + xv.w * xv.w;
  #pragma unroll
  for (int o = 16; o > 0; o >>= 1) {
    s  += __shfl_down(s, o, 32);
    s2 += __shfl_down(s2, o, 32);
  }
  if (lane == 0) { red[w] = s; red[8 + w] = s2; }
  __syncthreads();
  if (tid == 0) {
    float a = 0.f, c = 0.f;
    #pragma unroll
    for (int i = 0; i < 8; ++i) { a += red[i]; c += red[8 + i]; }
    float mean = a / (float)E_;
    red[0] = mean;
    red[1] = c / (float)E_ - mean * mean;
  }
  __syncthreads();
  float mean = red[0];
  float inv  = rsqrtf(red[1] + 1e-5f);
  float vv[4] = {xv.x, xv.y, xv.z, xv.w};
  #pragma unroll
  for (int j = 0; j < 4; ++j) {
    int c = tid * 4 + j;
    float n = (vv[j] - mean) * inv * g[c] + b[c];
    out[(size_t)row * E_ + c] = f2bf(n);
  }
}

// ------------------------------------------------------------ generic GEMM
// C[M,N] = A[M,K](bf16) * B[K,N](bf16)  (+bias)(relu)(+res f32) -> f32/bf16
// block tile 64(M) x 128(N), 8 waves (2x4), wave tile 32x32, K step 32.
// A tile staged by per-lane async global->LDS copies (ASYNCcnt),
// B tile staged by the Tensor Data Mover (TENSORcnt), one DMA per K step.
template <bool BIAS, bool RELU, bool RES, bool OUT32, bool OUT16>
__global__ __launch_bounds__(256) void gemm_bf16_kernel(
    const unsigned short* __restrict__ A, const unsigned short* __restrict__ Bm,
    const float* __restrict__ bias, const float* __restrict__ res,
    float* __restrict__ out32, unsigned short* __restrict__ out16,
    int M, int N, int K) {
  __shared__ unsigned short As[64 * 40];    // [64 rows][32 K] pad->40
#if USE_TDM
  __shared__ unsigned short Bt[32 * 128];   // row-major [32 K][128 N] (TDM)
#else
  __shared__ unsigned short Bt[128 * 40];   // transposed [128 N][32 K]
#endif

  const int tid  = threadIdx.x;
  const int lane = tid & 31, w = tid >> 5;
  const int l16 = lane & 15, half = lane >> 4;
  const int wm = w >> 2, wn = w & 3;
  const int m0 = blockIdx.y * 64;
  const int n0 = blockIdx.x * 128;

#if USE_TDM
  // D# group 1 (constant over the K loop): data_size=2B, tensor dim0=N,
  // dim0 stride=N, tile 128 x 32.
  v8u g1;
  g1[0] = 0x00010000u;                                   // data_size = 2 bytes
  g1[1] = ((unsigned)N & 0xFFFFu) << 16;                 // tensor_dim0 lo
  g1[2] = (((unsigned)K & 0xFFFFu) << 16) |
          (((unsigned)N >> 16) & 0xFFFFu);               // dim1 lo | dim0 hi
  g1[3] = (128u << 16) | (((unsigned)K >> 16) & 0xFFFFu);// tile_dim0 | dim1 hi
  g1[4] = 32u;                                           // tile_dim1
  g1[5] = (unsigned)N;                                   // dim0_stride lo
  g1[6] = 0u;
  g1[7] = 0u;
  const unsigned bt_base = lds_off(&Bt[0]);
#endif
#if USE_ASYNC
  const unsigned as_base = lds_off(&As[0]);
#endif

  v8f acc[2][2];
  #pragma unroll
  for (int mt = 0; mt < 2; ++mt)
    #pragma unroll
    for (int nt = 0; nt < 2; ++nt) acc[mt][nt] = vzero8();

  for (int k0 = 0; k0 < K; k0 += 32) {
#if USE_TDM
    if (w == 0) {                      // one DMA per block per K step
      unsigned long long ga =
          (unsigned long long)(const void*)&Bm[(size_t)k0 * N + n0];
      v4u g0;
      g0[0] = 1u;                                        // count=1
      g0[1] = bt_base;                                   // lds_addr
      g0[2] = (unsigned)ga;                              // global_addr lo
      g0[3] = ((unsigned)(ga >> 32) & 0x01FFFFFFu) | (2u << 30); // hi|type=2
      asm volatile("tensor_load_to_lds %0, %1" :: "s"(g0), "s"(g1) : "memory");
      wait_tensor0();
    }
#else
    #pragma unroll
    for (int i = 0; i < 2; ++i) {
      int idx = tid + i * 256;
      int kr = idx >> 4, nc = (idx & 15) * 8;
      uint4 bv = *(const uint4*)&Bm[(size_t)(k0 + kr) * N + n0 + nc];
      const unsigned short* pb = (const unsigned short*)&bv;
      #pragma unroll
      for (int j = 0; j < 8; ++j) Bt[(nc + j) * 40 + kr] = pb[j];
    }
#endif

    {
      int r = tid >> 2, c = (tid & 3) * 8;
      const unsigned short* gp = &A[(size_t)(m0 + r) * K + k0 + c];
#if USE_ASYNC
      async_g2l_b128(as_base + (unsigned)(r * 40 + c) * 2, gp);
#else
      *(uint4*)&As[r * 40 + c] = *(const uint4*)gp;
#endif
      if (k0 + 32 < K) {               // prefetch next B tile into L2
        __builtin_prefetch(&Bm[(size_t)(k0 + 32 + (tid >> 4)) * N + n0 +
                               (tid & 15) * 8], 0, 1);
      }
    }
#if USE_ASYNC
    wait_async0();
#endif
    __syncthreads();

    Frag af[2], bf[2];
    #pragma unroll
    for (int mt = 0; mt < 2; ++mt) {
      int rr = wm * 32 + mt * 16 + l16;
      af[mt].q[0] = *(const uint4*)&As[rr * 40 + half * 8];
      af[mt].q[1] = *(const uint4*)&As[rr * 40 + 16 + half * 8];
    }
    #pragma unroll
    for (int nt = 0; nt < 2; ++nt) {
      int cc = wn * 32 + nt * 16 + l16;
#if USE_TDM
      #pragma unroll
      for (int e = 0; e < 8; ++e) {                 // gather column cc
        bf[nt].us[e]     = Bt[(half * 8 + e) * 128 + cc];
        bf[nt].us[8 + e] = Bt[(16 + half * 8 + e) * 128 + cc];
      }
#else
      bf[nt].q[0] = *(const uint4*)&Bt[cc * 40 + half * 8];
      bf[nt].q[1] = *(const uint4*)&Bt[cc * 40 + 16 + half * 8];
#endif
    }
    #pragma unroll
    for (int mt = 0; mt < 2; ++mt)
      #pragma unroll
      for (int nt = 0; nt < 2; ++nt)
        acc[mt][nt] = wmma_bf16(af[mt], bf[nt], acc[mt][nt]);
    __syncthreads();
  }

  // epilogue; D layout: row = r + 8*half, col = l16
  #pragma unroll
  for (int mt = 0; mt < 2; ++mt)
    #pragma unroll
    for (int nt = 0; nt < 2; ++nt)
      #pragma unroll
      for (int r = 0; r < 8; ++r) {
        int gr = m0 + wm * 32 + mt * 16 + r + 8 * half;
        int gc = n0 + wn * 32 + nt * 16 + l16;
        float v = acc[mt][nt][r];
        if constexpr (BIAS) v += bias[gc];
        if constexpr (RELU) v = fmaxf(v, 0.f);
        if constexpr (RES)  v += res[(size_t)gr * N + gc];
        if constexpr (OUT32) out32[(size_t)gr * N + gc] = v;
        if constexpr (OUT16) out16[(size_t)gr * N + gc] = f2bf(v);
      }
}

// ---------------------------------------------------- attention pass 1:
// per-COLUMN (softmax over query axis t, as in the reference) max & sumexp.
// one wave per (bh, 16-column strip); scores = k[t]·k[s] * 1/32, causal t>=s.
__global__ __launch_bounds__(32) void attn_colstats_kernel(
    const unsigned short* __restrict__ kall,
    float* __restrict__ colmax, float* __restrict__ colsum) {
  const int lane = threadIdx.x;
  const int l16 = lane & 15, half = lane >> 4;
  const int sT = blockIdx.x;
  const int bh = blockIdx.y;
  const int b = bh >> 4, h = bh & 15;
  const int bt0 = b * T_;
  const int hoff = h * HS_;
  const int s_base = sT * 16;
  const float scale = 0.03125f;               // 1024^-0.5
  const float NINF = -__builtin_inff();

  Frag bf[2];                                 // key columns (loop-invariant)
  {
    const unsigned short* kp = &kall[(size_t)(bt0 + s_base + l16) * E_ + hoff];
    bf[0].q[0] = *(const uint4*)&kp[half * 8];
    bf[0].q[1] = *(const uint4*)&kp[16 + half * 8];
    bf[1].q[0] = *(const uint4*)&kp[32 + half * 8];
    bf[1].q[1] = *(const uint4*)&kp[48 + half * 8];
  }

  float m_run = NINF, s_run = 0.f;
  const int s = s_base + l16;

  for (int tT = sT; tT < T_ / 16; ++tT) {
    const int t_base = tT * 16;
    Frag af[2];
    const unsigned short* qp = &kall[(size_t)(bt0 + t_base + l16) * E_ + hoff];
    af[0].q[0] = *(const uint4*)&qp[half * 8];
    af[0].q[1] = *(const uint4*)&qp[16 + half * 8];
    af[1].q[0] = *(const uint4*)&qp[32 + half * 8];
    af[1].q[1] = *(const uint4*)&qp[48 + half * 8];

    v8f sc = vzero8();
    sc = wmma_bf16(af[0], bf[0], sc);
    sc = wmma_bf16(af[1], bf[1], sc);

    float vals[8];
    float tmax = NINF;
    #pragma unroll
    for (int r = 0; r < 8; ++r) {
      int t = t_base + r + 8 * half;
      if (t >= s) {
        float e = sc[r] * scale;
        vals[r] = e;
        tmax = fmaxf(tmax, e);
      } else {
        vals[r] = NINF;
      }
    }
    if (tmax > NINF) {
      float nm = fmaxf(m_run, tmax);
      float acc_s = (m_run > NINF) ? s_run * __expf(m_run - nm) : 0.f;
      #pragma unroll
      for (int r = 0; r < 8; ++r)
        if (vals[r] > NINF) acc_s += __expf(vals[r] - nm);
      s_run = acc_s;
      m_run = nm;
    }
  }
  // merge the two lane halves (same columns, different row groups)
  float om = __shfl_xor(m_run, 16, 32);
  float os = __shfl_xor(s_run, 16, 32);
  float nm = fmaxf(m_run, om);
  float tot = 0.f;
  if (m_run > NINF) tot += s_run * __expf(m_run - nm);
  if (om    > NINF) tot += os   * __expf(om    - nm);
  if (lane < 16) {
    colmax[(size_t)bh * T_ + s_base + lane] = nm;
    colsum[(size_t)bh * T_ + s_base + lane] = tot;
  }
}

// ---------------------------------------------------- attention pass 2:
// attn[t,d] = sum_s P[t,s] * v[s,d],  P = exp(score-colmax[s])/colsum[s].
// one wave per (bh, 16-row strip); P re-layouts through LDS, Vt staged in LDS.
__global__ __launch_bounds__(32) void attn_out_kernel(
    const unsigned short* __restrict__ kall,
    const unsigned short* __restrict__ vall,
    const float* __restrict__ colmax, const float* __restrict__ colsum,
    unsigned short* __restrict__ attnB) {
  __shared__ unsigned short Pl[16 * 40];     // P tile [16 t][32 s]
  __shared__ unsigned short Vt[64 * 40];     // V^T tile [64 d][32 s]

  const int lane = threadIdx.x;
  const int l16 = lane & 15, half = lane >> 4;
  const int tT = blockIdx.x;
  const int bh = blockIdx.y;
  const int b = bh >> 4, h = bh & 15;
  const int bt0 = b * T_;
  const int hoff = h * HS_;
  const int t_base = tT * 16;
  const float scale = 0.03125f;

  Frag qf[2];                                 // query rows (loop-invariant)
  {
    const unsigned short* qp = &kall[(size_t)(bt0 + t_base + l16) * E_ + hoff];
    qf[0].q[0] = *(const uint4*)&qp[half * 8];
    qf[0].q[1] = *(const uint4*)&qp[16 + half * 8];
    qf[1].q[0] = *(const uint4*)&qp[32 + half * 8];
    qf[1].q[1] = *(const uint4*)&qp[48 + half * 8];
  }
  v8f acc[4];
  #pragma unroll
  for (int i = 0; i < 4; ++i) acc[i] = vzero8();

  const int sBmax = (t_base + 15) >> 5;
  for (int sB = 0; sB <= sBmax; ++sB) {
    const int s0 = sB * 32;
    // stage V^T: 32 s-rows x 64 d -> Vt[d][s]
    #pragma unroll
    for (int i = 0; i < 8; ++i) {
      int c = lane + i * 32;
      int srow = c >> 3, dch = (c & 7) * 8;
      uint4 vv = *(const uint4*)&vall[(size_t)(bt0 + s0 + srow) * E_ + hoff + dch];
      const unsigned short* pv = (const unsigned short*)&vv;
      #pragma unroll
      for (int j = 0; j < 8; ++j) Vt[(dch + j) * 40 + srow] = pv[j];
    }
    // scores and normalized P for the two 16-column subtiles
    #pragma unroll
    for (int sh = 0; sh < 2; ++sh) {
      const int s_sub = s0 + sh * 16;
      Frag kf[2];
      const unsigned short* kp = &kall[(size_t)(bt0 + s_sub + l16) * E_ + hoff];
      kf[0].q[0] = *(const uint4*)&kp[half * 8];
      kf[0].q[1] = *(const uint4*)&kp[16 + half * 8];
      kf[1].q[0] = *(const uint4*)&kp[32 + half * 8];
      kf[1].q[1] = *(const uint4*)&kp[48 + half * 8];
      v8f sc = vzero8();
      sc = wmma_bf16(qf[0], kf[0], sc);
      sc = wmma_bf16(qf[1], kf[1], sc);

      const int s = s_sub + l16;
      float m  = colmax[(size_t)bh * T_ + s];
      float sm = colsum[(size_t)bh * T_ + s];
      float rs = (sm > 0.f) ? 1.f / sm : 0.f;
      #pragma unroll
      for (int r = 0; r < 8; ++r) {
        int t = t_base + r + 8 * half;
        float p = 0.f;
        if (s <= t && sm > 0.f) p = __expf(sc[r] * scale - m) * rs;
        Pl[(r + 8 * half) * 40 + sh * 16 + l16] = f2bf(p);
      }
    }
    __syncthreads();
    // attn += P @ V
    Frag pf;
    pf.q[0] = *(const uint4*)&Pl[l16 * 40 + half * 8];
    pf.q[1] = *(const uint4*)&Pl[l16 * 40 + 16 + half * 8];
    #pragma unroll
    for (int nt = 0; nt < 4; ++nt) {
      Frag vf;
      vf.q[0] = *(const uint4*)&Vt[(nt * 16 + l16) * 40 + half * 8];
      vf.q[1] = *(const uint4*)&Vt[(nt * 16 + l16) * 40 + 16 + half * 8];
      acc[nt] = wmma_bf16(pf, vf, acc[nt]);
    }
    __syncthreads();
  }

  // store heads concatenated: attnB[b*T+t][h*64+d]
  #pragma unroll
  for (int nt = 0; nt < 4; ++nt)
    #pragma unroll
    for (int r = 0; r < 8; ++r) {
      int t = t_base + r + 8 * half;
      int d = nt * 16 + l16;
      attnB[(size_t)(bt0 + t) * E_ + hoff + d] = f2bf(acc[nt][r]);
    }
}

// ---------------------------------------------------------------- launcher
extern "C" void kernel_launch(void* const* d_in, const int* in_sizes, int n_in,
                              void* d_out, int out_size, void* d_ws,
                              size_t ws_size, hipStream_t stream) {
  (void)in_sizes; (void)n_in; (void)out_size; (void)ws_size;

  const float* x      = (const float*)d_in[0];
  const float* ln1_g  = (const float*)d_in[1];
  const float* ln1_b  = (const float*)d_in[2];
  const float* Wk     = (const float*)d_in[3];
  const float* Wv     = (const float*)d_in[4];
  const float* Wproj  = (const float*)d_in[5];
  const float* bproj  = (const float*)d_in[6];
  const float* ln2_g  = (const float*)d_in[7];
  const float* ln2_b  = (const float*)d_in[8];
  const float* W1     = (const float*)d_in[9];
  const float* b1     = (const float*)d_in[10];
  const float* W2     = (const float*)d_in[11];
  const float* b2     = (const float*)d_in[12];
  float* out          = (float*)d_out;

  char* ws = (char*)d_ws;
  size_t off = 0;
  auto alloc = [&](size_t bytes) -> void* {
    void* p = ws + off;
    off += (bytes + 255) & ~(size_t)255;
    return p;
  };

  unsigned short* hB     = (unsigned short*)alloc((size_t)BT_ * E_ * 2);
  unsigned short* WkB    = (unsigned short*)alloc((size_t)E_ * E_ * 2);
  unsigned short* WvB    = (unsigned short*)alloc((size_t)E_ * E_ * 2);
  unsigned short* WprojB = (unsigned short*)alloc((size_t)E_ * E_ * 2);
  unsigned short* W1B    = (unsigned short*)alloc((size_t)E_ * FF_ * 2);
  unsigned short* W2B    = (unsigned short*)alloc((size_t)FF_ * E_ * 2);
  unsigned short* kall   = (unsigned short*)alloc((size_t)BT_ * E_ * 2);
  unsigned short* vall   = (unsigned short*)alloc((size_t)BT_ * E_ * 2);
  float*          cmax   = (float*)alloc((size_t)B_ * H_ * T_ * 4);
  float*          csum   = (float*)alloc((size_t)B_ * H_ * T_ * 4);
  unsigned short* attnB  = (unsigned short*)alloc((size_t)BT_ * E_ * 2);
  float*          x1     = (float*)alloc((size_t)BT_ * E_ * 4);
  unsigned short* h2B    = (unsigned short*)alloc((size_t)BT_ * E_ * 2);
  unsigned short* actB   = (unsigned short*)alloc((size_t)BT_ * FF_ * 2);

  // weight prep
  headperm_bf16_kernel<<<(E_ * E_) / 256, 256, 0, stream>>>(Wk, WkB);
  headperm_bf16_kernel<<<(E_ * E_) / 256, 256, 0, stream>>>(Wv, WvB);
  f32_to_bf16_kernel<<<(E_ * E_) / 256, 256, 0, stream>>>(Wproj, WprojB, E_ * E_);
  f32_to_bf16_kernel<<<(E_ * FF_) / 256, 256, 0, stream>>>(W1, W1B, E_ * FF_);
  f32_to_bf16_kernel<<<(FF_ * E_) / 256, 256, 0, stream>>>(W2, W2B, FF_ * E_);

  // LN1 -> h (bf16)
  layernorm_bf16_kernel<<<BT_, 256, 0, stream>>>(x, ln1_g, ln1_b, hB);

  // k = h @ Wk', v = h @ Wv'   (bf16 out for attention WMMA)
  dim3 gE(E_ / 128, BT_ / 64);
  gemm_bf16_kernel<false, false, false, false, true>
      <<<gE, 256, 0, stream>>>(hB, WkB, nullptr, nullptr, nullptr, kall,
                               BT_, E_, E_);
  gemm_bf16_kernel<false, false, false, false, true>
      <<<gE, 256, 0, stream>>>(hB, WvB, nullptr, nullptr, nullptr, vall,
                               BT_, E_, E_);

  // attention: column softmax stats, then output
  dim3 gA(T_ / 16, B_ * H_);
  attn_colstats_kernel<<<gA, 32, 0, stream>>>(kall, cmax, csum);
  attn_out_kernel<<<gA, 32, 0, stream>>>(kall, vall, cmax, csum, attnB);

  // x1 = x + attn @ Wproj + bproj   (f32)
  gemm_bf16_kernel<true, false, true, true, false>
      <<<gE, 256, 0, stream>>>(attnB, WprojB, bproj, x, x1, nullptr,
                               BT_, E_, E_);

  // LN2 -> h2 (bf16)
  layernorm_bf16_kernel<<<BT_, 256, 0, stream>>>(x1, ln2_g, ln2_b, h2B);

  // act = relu(h2 @ W1 + b1)   (bf16)
  dim3 gF(FF_ / 128, BT_ / 64);
  gemm_bf16_kernel<true, true, false, false, true>
      <<<gF, 256, 0, stream>>>(h2B, W1B, b1, nullptr, nullptr, actB,
                               BT_, FF_, E_);

  // out = x1 + act @ W2 + b2   (f32)
  gemm_bf16_kernel<true, false, true, true, false>
      <<<gE, 256, 0, stream>>>(actB, W2B, b2, x1, out, nullptr,
                               BT_, E_, FF_);
}
